// MLPPredictor_70162585748136
// MI455X (gfx1250) — matrix operand order repaired and checked by reading, not compiled, not run on hardware
//
#include <hip/hip_runtime.h>

typedef __attribute__((ext_vector_type(16))) _Float16 v16h;
typedef __attribute__((ext_vector_type(8)))  float    v8f;

#define N_EDGES  600000
#define N_TILES  (N_EDGES / 16)   // 37500, exact
#define IN_FEATS 128
#define HIDDEN   128
#define KDIM     256              // 2 * IN_FEATS

union A16 { v16h v; _Float16 e[16]; };
union F8  { v8f  v; float    e[8];  };

__global__ __launch_bounds__(256) void edge_mlp_wmma_kernel(
    const int* __restrict__ src, const int* __restrict__ dst,
    const float* __restrict__ h,  const float* __restrict__ W1,
    const float* __restrict__ b1, const float* __restrict__ W2,
    const float* __restrict__ b2, float* __restrict__ out)
{
    // W1 staged as f16, row-major [HIDDEN][KDIM] = 64 KB LDS (WGP has 320 KB)
    __shared__ _Float16 sW1[HIDDEN * KDIM];

    const int tid = threadIdx.x;

    // Stage W1: 32768 floats -> f16. 256 threads x 32 float4 each.
    {
        const float4* w4 = (const float4*)W1;
        #pragma unroll 4
        for (int i = tid * 32; i < tid * 32 + 32; ++i) {
            float4 v = w4[i];
            sW1[4 * i + 0] = (_Float16)v.x;
            sW1[4 * i + 1] = (_Float16)v.y;
            sW1[4 * i + 2] = (_Float16)v.z;
            sW1[4 * i + 3] = (_Float16)v.w;
        }
    }
    __syncthreads();

    const int lane        = tid & 31;
    const int col         = lane & 15;   // matrix column / A row within tile
    const int hi          = lane >> 4;   // half-wave selector
    const int waveInBlk   = tid >> 5;
    const int wavesPerBlk = blockDim.x >> 5;
    const int globalWave  = blockIdx.x * wavesPerBlk + waveInBlk;
    const int totalWaves  = gridDim.x * wavesPerBlk;

    const float b2v = b2[0];
    const v8f zero = {0.f, 0.f, 0.f, 0.f, 0.f, 0.f, 0.f, 0.f};

    for (int t = globalWave; t < N_TILES; t += totalWaves) {
        // A-matrix row for this lane: M = col; edge index:
        const int e  = t * 16 + col;
        const int si = src[e];
        const int di = dst[e];
        const float* srow = h + (size_t)si * IN_FEATS;
        const float* drow = h + (size_t)di * IN_FEATS;

        // fc2 per-row partials, persistent across both passes.
        float partial[8];
        #pragma unroll
        for (int j = 0; j < 8; ++j) partial[j] = 0.f;

        // Two passes of 4 N-tiles each: keeps live accumulators at 32 VGPRs
        // (8 x v8f spilled to scratch in the previous version).
        #pragma unroll 1
        for (int pass = 0; pass < 2; ++pass) {
            const int nt0 = pass * 4;

            v8f acc[4];
            #pragma unroll
            for (int q = 0; q < 4; ++q) acc[q] = zero;

            #pragma unroll
            for (int ks = 0; ks < 8; ++ks) {
                const int k0 = ks * 32;
                // concat(h[src], h[dst]): K<128 -> src row, K>=128 -> dst row
                const float* base = (k0 < IN_FEATS) ? (srow + k0)
                                                    : (drow + (k0 - IN_FEATS));
                const int kb = hi * 16; // lanes 0-15: K=0..15; 16-31: K=16..31

                const float4 f0 = *(const float4*)(base + kb);
                const float4 f1 = *(const float4*)(base + kb + 4);
                const float4 f2 = *(const float4*)(base + kb + 8);
                const float4 f3 = *(const float4*)(base + kb + 12);

                A16 a;
                a.e[0]  = (_Float16)f0.x; a.e[1]  = (_Float16)f0.y;
                a.e[2]  = (_Float16)f0.z; a.e[3]  = (_Float16)f0.w;
                a.e[4]  = (_Float16)f1.x; a.e[5]  = (_Float16)f1.y;
                a.e[6]  = (_Float16)f1.z; a.e[7]  = (_Float16)f1.w;
                a.e[8]  = (_Float16)f2.x; a.e[9]  = (_Float16)f2.y;
                a.e[10] = (_Float16)f2.z; a.e[11] = (_Float16)f2.w;
                a.e[12] = (_Float16)f3.x; a.e[13] = (_Float16)f3.y;
                a.e[14] = (_Float16)f3.z; a.e[15] = (_Float16)f3.w;

                const int kb16 = hi * 16; // B: lanes 0-15 K=0..15, lanes 16-31 K=16..31
                #pragma unroll
                for (int q = 0; q < 4; ++q) {
                    const int r = (nt0 + q) * 16 + col;   // W1 output row = B column
                    const v16h bm = *(const v16h*)(sW1 + r * KDIM + k0 + kb16);
                    acc[q] = __builtin_amdgcn_wmma_f32_16x16x32_f16(
                        /*neg_a=*/false, a.v, /*neg_b=*/false, bm,
                        /*c_mod=*/(short)0, acc[q], /*reuse_a=*/false, /*reuse_b=*/false);
                }
            }

            // Fold this pass's 4 N-tiles into the fc2 partials immediately so
            // the accumulators die here.
            #pragma unroll
            for (int q = 0; q < 4; ++q) {
                const int nt = nt0 + q;
                const float b1q = b1[nt * 16 + col];
                const float w2q = W2[nt * 16 + col];
                F8 c; c.v = acc[q];
                #pragma unroll
                for (int j = 0; j < 8; ++j) {
                    float hv = c.e[j] + b1q;
                    hv = fmaxf(hv, 0.f);                  // ReLU
                    partial[j] = fmaf(hv, w2q, partial[j]);
                }
            }
        }

        // Reduce the 16 columns within each half-wave (xor masks stay in-half).
        // C/D layout: acc element j, lane L -> row M = (L>>4)*8 + j, col N = L&15.
        #pragma unroll
        for (int j = 0; j < 8; ++j) {
            #pragma unroll
            for (int off = 1; off < 16; off <<= 1)
                partial[j] += __shfl_xor(partial[j], off, 32);
        }

        if (col == 0) {
            const int mbase = t * 16 + hi * 8;
            #pragma unroll
            for (int j = 0; j < 8; ++j) {
                float x = partial[j] + b2v;
                out[mbase + j] = 1.f / (1.f + __expf(-x)); // sigmoid
            }
        }
    }
}

extern "C" void kernel_launch(void* const* d_in, const int* in_sizes, int n_in,
                              void* d_out, int out_size, void* d_ws, size_t ws_size,
                              hipStream_t stream) {
    const int*   src = (const int*)  d_in[0];
    const int*   dst = (const int*)  d_in[1];
    const float* h   = (const float*)d_in[2];
    const float* W1  = (const float*)d_in[3];
    const float* b1  = (const float*)d_in[4];
    const float* W2  = (const float*)d_in[5];
    const float* b2  = (const float*)d_in[6];
    float* out = (float*)d_out;

    dim3 grid(640);    // 640 blocks x 8 waves = 5120 waves; ~7.3 tiles each
    dim3 block(256);   // 8 wave32 per block share the LDS-staged W1
    hipLaunchKernelGGL(edge_mlp_wmma_kernel, grid, block, 0, stream,
                       src, dst, h, W1, b1, W2, b2, out);
}
